// EmbeddingWeightedConv_75634374082660
// MI455X (gfx1250) — matrix operand order repaired and checked by reading, not compiled, not run on hardware
//
#include <hip/hip_runtime.h>

// Problem constants (from reference): N=32, E=512, C=64, O=128, H=W=64, K=3
#define NN   32
#define EE   512
#define CC   64
#define OO   128
#define HH   64
#define WWD  64
#define KK9  9            // 3x3 taps
#define CKK  576          // C * 9
#define OCKK (OO * CKK)   // 73728 filter elements per sample
#define PH   66           // padded height (1-px zero border)
#define PW   66           // padded width

typedef __attribute__((ext_vector_type(16))) __bf16 v16bf;
typedef __attribute__((ext_vector_type(8)))  float  v8f;

union BfPack { uint4 q[2]; v16bf v; };
union F16Pack { float4 f4[4]; float s[16]; };

// --- CDNA5 async global->LDS copy (ASYNCcnt-tracked), via inline asm --------
__device__ __forceinline__ void async_b128_to_lds(uint32_t lds_byte_addr,
                                                  const void* gptr) {
    asm volatile("global_load_async_to_lds_b128 %0, %1, off"
                 :: "v"(lds_byte_addr), "v"((unsigned long long)(uintptr_t)gptr)
                 : "memory");
}
__device__ __forceinline__ void wait_async0() {
    asm volatile("s_wait_asynccnt 0x0" ::: "memory");
}

// ---------------------------------------------------------------------------
// Kernel 1: embedding f32 -> bf16 (tiny, 16K elements)
// ---------------------------------------------------------------------------
__global__ __launch_bounds__(256) void cvt_emb_kernel(const float* __restrict__ e,
                                                      __bf16* __restrict__ ebf) {
    int i = blockIdx.x * 256 + threadIdx.x;
    if (i < NN * EE) ebf[i] = (__bf16)e[i];
}

// ---------------------------------------------------------------------------
// Kernel 2: c1_b[n,o] = emb[n,:] . b_weight[o,:] + b_bias[o]   (tiny GEMM)
// ---------------------------------------------------------------------------
__global__ __launch_bounds__(256) void gen_bias_kernel(const float* __restrict__ emb,
                                                       const float* __restrict__ bw,
                                                       const float* __restrict__ bb,
                                                       float* __restrict__ cb) {
    int id = blockIdx.x * 256 + threadIdx.x;     // 32*128 = 4096
    if (id >= NN * OO) return;
    int n = id >> 7, o = id & 127;
    float s = bb[o];
    const float* ep = emb + (size_t)n * EE;
    const float* wp = bw + (size_t)o * EE;
    for (int k = 0; k < EE; k += 4) {
        float4 a = *(const float4*)(ep + k);
        float4 b = *(const float4*)(wp + k);
        s = fmaf(a.x, b.x, s); s = fmaf(a.y, b.y, s);
        s = fmaf(a.z, b.z, s); s = fmaf(a.w, b.w, s);
    }
    cb[id] = s;
}

// ---------------------------------------------------------------------------
// Kernel 3a: zero the padded NHWC image buffer (borders stay zero)
// ---------------------------------------------------------------------------
__global__ __launch_bounds__(256) void zero_pimg_kernel(uint4* __restrict__ p) {
    // NN*PH*PW*CC bf16 = 17,842,176 B = 1,115,136 uint4
    size_t i = (size_t)blockIdx.x * 256 + threadIdx.x;
    if (i < (size_t)NN * PH * PW * CC / 8) {
        uint4 z; z.x = 0u; z.y = 0u; z.z = 0u; z.w = 0u;
        p[i] = z;
    }
}

// ---------------------------------------------------------------------------
// Kernel 3b: NCHW f32 -> padded NHWC bf16 via LDS tile transpose.
//   One block per (n, y): transpose the 64(c) x 64(x) slab, coalesced on
//   both the read (x-contiguous) and write (c-contiguous) sides.
// ---------------------------------------------------------------------------
__global__ __launch_bounds__(256) void transpose_img_kernel(
    const float* __restrict__ images, __bf16* __restrict__ pimg) {
    __shared__ float t[64 * 65];                 // 65 stride: no bank conflicts
    const int n = blockIdx.x, y = blockIdx.y;
    const int tid = threadIdx.x;
    for (int idx = tid; idx < 4096; idx += 256) {
        const int c = idx >> 6, x = idx & 63;
        t[c * 65 + x] = images[(((size_t)n * CC + c) * HH + y) * WWD + x];
    }
    __syncthreads();
    for (int idx = tid; idx < 4096; idx += 256) {
        const int x = idx >> 6, c = idx & 63;
        pimg[(((size_t)n * PH + (y + 1)) * PW + (x + 1)) * CC + c] = (__bf16)t[c * 65 + x];
    }
}

// ---------------------------------------------------------------------------
// Kernel 4: filter generation GEMM with WMMA bf16 (the 151 MB HBM stream).
//   fw[n, o*576 + pos*64 + c] = sum_k emb[n,k] * w_weight[o*576 + c*9 + pos, k]
//                               + w_bias[...]
//   K index reordered: tap-position-major, channel-fastest -> conv-friendly.
//   One wave owns a 32x16 output tile: 2 f32 accumulators, 32 WMMAs.
// ---------------------------------------------------------------------------
__global__ __launch_bounds__(256) void gen_filters_kernel(
    const __bf16* __restrict__ embbf, const float* __restrict__ w_weight,
    const float* __restrict__ w_bias, __bf16* __restrict__ fw) {

    const int tid  = threadIdx.x;
    const int lane = tid & 31;
    const int gw   = blockIdx.x * 8 + (tid >> 5);   // global wave id, 4608 total
    const int col0 = gw * 16;                       // reordered column base
    const int o    = col0 / CKK;
    const int rem  = col0 - o * CKK;
    const int pos  = rem >> 6;                      // tap position 0..8
    const int c0   = rem & 63;                      // channel base (0/16/32/48)
    const int half = lane >> 4;
    const int lm   = lane & 15;
    const int cidx = c0 + lm;                       // this lane's B column channel
    const size_t wrow = (size_t)(o * CKK + cidx * KK9 + pos) * EE;

    v8f acc0 = {};   // samples 0..15
    v8f acc1 = {};   // samples 16..31

    for (int kt = 0; kt < 16; ++kt) {
        const int k0 = kt * 32;

        // A tiles (embedding, bf16): lane m=L%16 holds K = half*8 + {0..7}
        // (V0-3) and half*8 + 16 + {0..7} (V4-7).
        BfPack A0, A1;
        const __bf16* e0 = embbf + (size_t)lm * EE + k0 + half * 8;
        A0.q[0] = *(const uint4*)(e0);
        A0.q[1] = *(const uint4*)(e0 + 16);
        const __bf16* e1 = e0 + (size_t)16 * EE;
        A1.q[0] = *(const uint4*)(e1);
        A1.q[1] = *(const uint4*)(e1 + 16);

        // B tile: lane col = L%16, K = half*8 + {0..7} and half*8 + 16 + {0..7}; f32 -> bf16.
        const float* wp = w_weight + wrow + k0 + half * 8;
        if (kt < 15) __builtin_prefetch(wp + 32, 0, 1);
        F16Pack fp;
        fp.f4[0] = ((const float4*)wp)[0];
        fp.f4[1] = ((const float4*)wp)[1];
        fp.f4[2] = ((const float4*)(wp + 16))[0];
        fp.f4[3] = ((const float4*)(wp + 16))[1];
        v16bf B;
        #pragma unroll
        for (int i = 0; i < 16; ++i) B[i] = (__bf16)fp.s[i];

        acc0 = __builtin_amdgcn_wmma_f32_16x16x32_bf16(false, A0.v, false, B,
                                                       (short)0, acc0, false, false);
        acc1 = __builtin_amdgcn_wmma_f32_16x16x32_bf16(false, A1.v, false, B,
                                                       (short)0, acc1, false, false);
    }

    const float bias = w_bias[o * CKK + cidx * KK9 + pos];
    // D layout: VGPR r, lanes 0-15 -> M=r, lanes 16-31 -> M=r+8; N = L%16.
    #pragma unroll
    for (int r = 0; r < 8; ++r) {
        const int m = r + half * 8;
        fw[(size_t)m * OCKK + col0 + lm]        = (__bf16)(acc0[r] + bias);
        fw[(size_t)(m + 16) * OCKK + col0 + lm] = (__bf16)(acc1[r] + bias);
    }
}

// ---------------------------------------------------------------------------
// Kernel 5: dynamic conv as per-sample implicit GEMM with WMMA bf16.
//   Block = (sample n, 4x16 pixel tile, 64-output group).
//   Halo tile (6 x 18 x 64ch, bf16, [y][x][c]) staged from the padded NHWC
//   image with ASYNC global->LDS b128 copies (contiguous 2304B per row; zero
//   border means no bounds checks). 8 waves = 4 o-subtiles x 2 pixel pairs,
//   2 accumulators/wave, 18 K-steps -> 36 WMMAs/wave.
// ---------------------------------------------------------------------------
__global__ __launch_bounds__(256) void dyn_conv_kernel(
    const __bf16* __restrict__ fw, const __bf16* __restrict__ pimg,
    const float* __restrict__ cb, float* __restrict__ out) {

    __shared__ __bf16 simg[6 * 18 * 64];         // 13.8 KB halo tile, [y][x][c]

    const int n  = blockIdx.x;
    const int ty = blockIdx.y >> 2, tx = blockIdx.y & 3;
    const int og = blockIdx.z;
    const int y0 = ty * 4, x0 = tx * 16;
    const int tid = threadIdx.x;

    // Async-stage 6 contiguous rows of 18*64 bf16 (2304 B each) into LDS.
    // Padded image: image row (y0-1+yy) -> padded row (y0+yy), col x0-1 -> x0.
    {
        const __bf16* src = pimg + (((size_t)n * PH + y0) * PW + x0) * CC;
        const uint32_t lds0 = (uint32_t)(uintptr_t)(&simg[0]);
        for (int idx = tid; idx < 864; idx += 256) {   // 864 x 16B chunks
            const int row = idx / 144;
            const int off = idx - row * 144;
            async_b128_to_lds(lds0 + (uint32_t)(row * 2304 + off * 16),
                              (const void*)(src + (size_t)row * PW * CC + off * 8));
        }
        wait_async0();
    }
    __syncthreads();

    const int wave = tid >> 5, lane = tid & 31;
    const int mt = wave & 3, nt = wave >> 2;
    const int o0 = og * 64 + mt * 16;
    const int half = lane >> 4, lm = lane & 15;
    const int orow = o0 + lm;                    // A-operand row for this lane

    // Pixel rows for the two accumulators (16 lanes = 16 consecutive x).
    const int py0 = nt * 2 + 0;
    const int py1 = nt * 2 + 1;

    v8f acc0 = {};
    v8f acc1 = {};
    const __bf16* fbase = fw + (size_t)n * OCKK + (size_t)orow * CKK;

    for (int kt = 0; kt < 18; ++kt) {
        const int pos   = kt >> 1;               // tap position 0..8
        const int cbase = (kt & 1) * 32;         // channel half
        const int ti = pos / 3, tj = pos - ti * 3;

        // A: filters (bf16, reordered layout -> contiguous 16B chunks)
        BfPack A;
        const __bf16* fa = fbase + pos * 64 + cbase + half * 8;
        A.q[0] = *(const uint4*)(fa);
        A.q[1] = *(const uint4*)(fa + 16);

        // B: channels cbase + half*8 + {0..7, 16..23} at the shifted pixel, straight from LDS
        BfPack B0, B1;
        const __bf16* pb0 = &simg[((py0 + ti) * 18 + (lm + tj)) * 64 + cbase + half * 8];
        B0.q[0] = *(const uint4*)(pb0);
        B0.q[1] = *(const uint4*)(pb0 + 16);
        const __bf16* pb1 = &simg[((py1 + ti) * 18 + (lm + tj)) * 64 + cbase + half * 8];
        B1.q[0] = *(const uint4*)(pb1);
        B1.q[1] = *(const uint4*)(pb1 + 16);

        acc0 = __builtin_amdgcn_wmma_f32_16x16x32_bf16(false, A.v, false, B0.v,
                                                       (short)0, acc0, false, false);
        acc1 = __builtin_amdgcn_wmma_f32_16x16x32_bf16(false, A.v, false, B1.v,
                                                       (short)0, acc1, false, false);
    }

    // Epilogue: add generated bias (varies along M => per-VGPR), store f32.
    // 16 lanes of a D row cover 16 consecutive x -> 64B contiguous stores.
    #pragma unroll
    for (int r = 0; r < 8; ++r) {
        const int o = o0 + r + half * 8;
        const float bv = cb[n * OO + o];
        out[(((size_t)n * OO + o) * HH + (y0 + py0)) * WWD + (x0 + lm)] = acc0[r] + bv;
        out[(((size_t)n * OO + o) * HH + (y0 + py1)) * WWD + (x0 + lm)] = acc1[r] + bv;
    }
}

// ---------------------------------------------------------------------------
extern "C" void kernel_launch(void* const* d_in, const int* in_sizes, int n_in,
                              void* d_out, int out_size, void* d_ws, size_t ws_size,
                              hipStream_t stream) {
    (void)in_sizes; (void)n_in; (void)out_size; (void)ws_size;
    const float* embedding = (const float*)d_in[0];   // [32, 512]
    const float* images    = (const float*)d_in[1];   // [32, 64, 64, 64]
    const float* w_weight  = (const float*)d_in[2];   // [73728, 512]
    const float* w_bias    = (const float*)d_in[3];   // [73728]
    const float* b_weight  = (const float*)d_in[4];   // [128, 512]
    const float* b_bias    = (const float*)d_in[5];   // [128]
    float* out = (float*)d_out;                       // [32, 128, 64, 64]

    // Workspace carve-up (~22.6 MB total)
    char* ws = (char*)d_ws;
    size_t off = 0;
    __bf16* fw  = (__bf16*)(ws + off); off += (size_t)NN * OCKK * 2;       // 4.72 MB
    __bf16* ebf = (__bf16*)(ws + off); off += (size_t)NN * EE * 2;         // 32 KB
    float*  cbb = (float*)(ws + off);  off += (size_t)NN * OO * 4;         // 16 KB
    __bf16* pim = (__bf16*)(ws + off);                                     // 17.0 MB

    cvt_emb_kernel<<<64, 256, 0, stream>>>(embedding, ebf);
    gen_bias_kernel<<<16, 256, 0, stream>>>(embedding, b_weight, b_bias, cbb);

    const size_t pimg_chunks = (size_t)NN * PH * PW * CC / 8;   // uint4 count
    zero_pimg_kernel<<<(unsigned)((pimg_chunks + 255) / 256), 256, 0, stream>>>((uint4*)pim);
    dim3 tg(NN, HH);
    transpose_img_kernel<<<tg, 256, 0, stream>>>(images, pim);

    gen_filters_kernel<<<576, 256, 0, stream>>>(ebf, w_weight, w_bias, fw);

    dim3 grid(NN, 64, 2);
    dyn_conv_kernel<<<grid, 256, 0, stream>>>(fw, pim, cbb, out);
}
